// MaskGAT_56977036149415
// MI455X (gfx1250) — matrix operand order repaired and checked by reading, not compiled, not run on hardware
//
#include <hip/hip_runtime.h>
#include <hip/hip_bf16.h>

#define N_NODES 2048
#define N_EDGES 32768
#define DIM     512
#define DIM2    1024
#define TOPK    10
#define NEGV    (-1e15f)

typedef __attribute__((ext_vector_type(16))) __bf16 v16bf;
typedef __attribute__((ext_vector_type(8)))  __bf16 v8bf;
typedef __attribute__((ext_vector_type(8)))  float  v8f;

// 128-bit payload type exactly matching the async-LDS builtin's prototype
// (clang expects: 'int __vector(4) __device__ *' / '... __shared__ *')
typedef int v4i_ls __attribute__((vector_size(16)));
#if defined(__HIP_DEVICE_COMPILE__)
typedef __attribute__((address_space(1))) v4i_ls* g_b128_ptr;
typedef __attribute__((address_space(3))) v4i_ls* l_b128_ptr;
#endif

// ---------------------------------------------------------------------------
// f32 -> bf16 conversion
// ---------------------------------------------------------------------------
__global__ void cvt_bf16_kernel(const float* __restrict__ in,
                                unsigned short* __restrict__ out, int n) {
  int i = blockIdx.x * 256 + threadIdx.x;
  if (i < n) ((__bf16*)out)[i] = (__bf16)in[i];
}

__global__ void zero_kernel(float* __restrict__ p, int n) {
  int i = blockIdx.x * 256 + threadIdx.x;
  if (i < n) p[i] = 0.f;
}

// ---------------------------------------------------------------------------
// Edge-gather GEMM:  m[e, o] = sum_k concat(X1[idx1[e]], X2[idx2[e]])[k] * W[o,k] + b[o]
// Block = 16 edges x 128 outputs (8 waves, one 16x16 WMMA tile each).
// The 16 gathered/concatenated A rows (16 x 1024 bf16 = 32 KB) are staged once
// into LDS via async global->LDS DMA (ASYNCcnt / s_wait_asynccnt), shared by
// all 8 waves; B (W^T columns) streams from global (L2-resident) per wave.
// A frag (16x32 bf16): lanes 0-15 row M=lane, K k..k+15;
//                      lanes 16-31 row M=lane-16, K k+16..k+31.
// B frag (32x16 bf16): lanes 0-15 col N=lane, K k..k+15; lanes 16-31 K k+16..k+31.
// ---------------------------------------------------------------------------
__global__ __launch_bounds__(256)
void gat_gemm_bf16_kernel(const unsigned short* __restrict__ X1u, const int* __restrict__ idx1,
                          const unsigned short* __restrict__ X2u, const int* __restrict__ idx2,
                          const unsigned short* __restrict__ Wu,  const float* __restrict__ bias,
                          float* __restrict__ out) {
  const __bf16* X1 = (const __bf16*)X1u;
  const __bf16* X2 = (const __bf16*)X2u;
  const __bf16* W  = (const __bf16*)Wu;

  __shared__ __align__(16) __bf16 smA[16 * DIM2];   // 32 KB: 16 edge rows x 1024

  const int tid   = threadIdx.x;
  const int lane  = tid & 31;
  const int wave  = tid >> 5;
  const int etile = blockIdx.x;                 // edge tile (16 edges)
  const int otile = blockIdx.y * 8 + wave;      // output tile (16 outs)
  const int mrow  = lane & 15;
  const int kgrp  = lane >> 4;                  // 0 or 1

  // ---- stage 16 concatenated rows into LDS: 2048 x 16B chunks, 8 per thread
#pragma unroll
  for (int it = 0; it < 8; ++it) {
    const int c     = it * 256 + tid;           // chunk id [0, 2048)
    const int row   = c >> 7;                   // 128 chunks per row
    const int kelem = (c & 127) * 8;            // bf16 offset within 1024
    const int eg    = etile * 16 + row;
    const __bf16* g;
    if (kelem < DIM) {
      g = X1 + (size_t)idx1[eg] * DIM + kelem;
    } else {
      const int r2 = idx2 ? idx2[eg] : eg;
      g = X2 + (size_t)r2 * DIM + (kelem - DIM);
    }
    __bf16* l = &smA[row * DIM2 + kelem];
#if defined(__HIP_DEVICE_COMPILE__) && \
    __has_builtin(__builtin_amdgcn_global_load_async_to_lds_b128)
    __builtin_amdgcn_global_load_async_to_lds_b128(
        (g_b128_ptr)(void*)g, (l_b128_ptr)(void*)l, 0, 0);
#else
    *(v8bf*)l = *(const v8bf*)g;
#endif
  }
#if defined(__HIP_DEVICE_COMPILE__) && \
    __has_builtin(__builtin_amdgcn_global_load_async_to_lds_b128)
#if __has_builtin(__builtin_amdgcn_s_wait_asynccnt)
  __builtin_amdgcn_s_wait_asynccnt(0);
#else
  asm volatile("s_wait_asynccnt 0x0" ::: "memory");
#endif
#endif
  __syncthreads();

  // ---- WMMA K-loop: A from LDS, B from global
  const int o = otile * 16 + mrow;              // B column for this lane
  const __bf16* arow = &smA[mrow * DIM2];
  const __bf16* bw   = W + (size_t)o * DIM2;

  v8f acc = {};
  for (int k = 0; k < DIM2; k += 32) {
    const int kb = k + kgrp * 16;
    const __bf16* bbase = bw + k + kgrp * 16;
    __builtin_prefetch(bbase + 32, 0, 1);       // global_prefetch_b8 for next K chunk
    v8bf alo = *(const v8bf*)(arow + kb);       // ds_load_b128
    v8bf ahi = *(const v8bf*)(arow + kb + 8);   // ds_load_b128
    v8bf blo = *(const v8bf*)(bbase);
    v8bf bhi = *(const v8bf*)(bbase + 8);
    v16bf a = __builtin_shufflevector(alo, ahi, 0,1,2,3,4,5,6,7,8,9,10,11,12,13,14,15);
    v16bf b = __builtin_shufflevector(blo, bhi, 0,1,2,3,4,5,6,7,8,9,10,11,12,13,14,15);
    acc = __builtin_amdgcn_wmma_f32_16x16x32_bf16(false, a, false, b, (short)0, acc,
                                                  false, false);
  }

  const float bo = bias[o];
#pragma unroll
  for (int r = 0; r < 8; ++r) {
    const int erow = etile * 16 + r + 8 * kgrp; // C/D layout: M = r (+8 for hi lanes)
    out[(size_t)erow * DIM + o] = acc[r] + bo;
  }
}

// ---------------------------------------------------------------------------
// Attention score: e[e] = leaky_relu(dot(m[e,:], aW) + ab, 0.2). One wave/edge.
// ---------------------------------------------------------------------------
__global__ void attn_score_kernel(const float* __restrict__ m, const float* __restrict__ aW,
                                  const float* __restrict__ ab, float* __restrict__ eo) {
  int gw   = (blockIdx.x * blockDim.x + threadIdx.x) >> 5;
  int lane = threadIdx.x & 31;
  if (gw >= N_EDGES) return;
  const float* row = m + (size_t)gw * DIM;
  float s = 0.f;
#pragma unroll
  for (int i = 0; i < DIM / 32; ++i) s += row[lane + i * 32] * aW[lane + i * 32];
#pragma unroll
  for (int off = 16; off > 0; off >>= 1) s += __shfl_xor(s, off, 32);
  if (lane == 0) {
    float x = s + ab[0];
    eo[gw] = (x > 0.f) ? x : 0.2f * x;
  }
}

// ---------------------------------------------------------------------------
// Segment max + exp-sum per node (block = node, scan edge list).
// ---------------------------------------------------------------------------
__global__ void seg_reduce_kernel(const float* __restrict__ e, const int* __restrict__ idx,
                                  float* __restrict__ smax, float* __restrict__ ssum) {
  const int node = blockIdx.x;
  const int tid  = threadIdx.x;
  __shared__ float red[256];

  float mx = -3.0e38f;
  for (int i = tid; i < N_EDGES; i += 256)
    if (idx[i] == node) mx = fmaxf(mx, e[i]);
  red[tid] = mx;
  __syncthreads();
  for (int s = 128; s > 0; s >>= 1) {
    if (tid < s) red[tid] = fmaxf(red[tid], red[tid + s]);
    __syncthreads();
  }
  const float m0 = red[0];
  __syncthreads();

  float sm = 0.f;
  for (int i = tid; i < N_EDGES; i += 256)
    if (idx[i] == node) sm += __expf(e[i] - m0);
  red[tid] = sm;
  __syncthreads();
  for (int s = 128; s > 0; s >>= 1) {
    if (tid < s) red[tid] += red[tid + s];
    __syncthreads();
  }
  if (tid == 0) { smax[node] = m0; ssum[node] = red[0]; }
}

// ---------------------------------------------------------------------------
// Purifier: per group (row for j=1, col for j=0) find 10th largest score.
// Dense mask rows are NEG-filled, so groups with <10 edges -> thresh = NEG.
// ---------------------------------------------------------------------------
__global__ void topk_thresh_kernel(const float* __restrict__ ev, const int* __restrict__ idx,
                                   float* __restrict__ thr) {
  const int node = blockIdx.x;
  __shared__ float vals[2048];
  __shared__ int cnt;
  if (threadIdx.x == 0) cnt = 0;
  __syncthreads();
  for (int i = threadIdx.x; i < N_EDGES; i += 256)
    if (idx[i] == node) {
      int p = atomicAdd(&cnt, 1);
      if (p < 2048) vals[p] = ev[i];
    }
  __syncthreads();
  if (threadIdx.x == 0) {
    int c = cnt < 2048 ? cnt : 2048;
    if (c < TOPK) {
      thr[node] = NEGV;
    } else {
      float th = NEGV;
      for (int t = 0; t < TOPK; ++t) {       // selection with index marking
        int bi = 0; float bv = -3.0e38f;
        for (int j = 0; j < c; ++j)
          if (vals[j] > bv) { bv = vals[j]; bi = j; }
        th = bv; vals[bi] = -3.0e38f;
      }
      thr[node] = th;
    }
  }
}

__global__ void apply_thresh_kernel(const float* __restrict__ e, const int* __restrict__ idx,
                                    const float* __restrict__ thr, float* __restrict__ e2) {
  int i = blockIdx.x * 256 + threadIdx.x;
  if (i < N_EDGES) {
    float v = e[i];
    e2[i] = (v <= thr[idx[i]]) ? NEGV : v;   // where(mask <= thresh, NEG, mask)
  }
}

// ---------------------------------------------------------------------------
// rel_emb accumulation: out = (init ? ef : out) + 0.5 * coef(e) * m
// ---------------------------------------------------------------------------
__global__ void rel_accum_kernel(const float* __restrict__ m, const float* __restrict__ e,
                                 const float* __restrict__ smax, const float* __restrict__ ssum,
                                 const int* __restrict__ idx, const float* __restrict__ ef,
                                 float* __restrict__ out, int init) {
  size_t i = (size_t)blockIdx.x * 256 + threadIdx.x;
  if (i >= (size_t)N_EDGES * DIM) return;
  int ed   = (int)(i >> 9);
  int node = idx[ed];
  float coef = __expf(e[ed] - smax[node]) / (ssum[node] + 1e-16f);
  float v = 0.5f * coef * m[i];
  out[i] = init ? (ef[i] + v) : (out[i] + v);
}

// ---------------------------------------------------------------------------
// Node scatter: acc[idx[e], :] += coef(e) * m[e, :]
// ---------------------------------------------------------------------------
__global__ void node_accum_kernel(const float* __restrict__ m, const float* __restrict__ e2,
                                  const float* __restrict__ smax, const float* __restrict__ ssum,
                                  const int* __restrict__ idx, float* __restrict__ acc) {
  size_t i = (size_t)blockIdx.x * 256 + threadIdx.x;
  if (i >= (size_t)N_EDGES * DIM) return;
  int ed   = (int)(i >> 9);
  int d    = (int)(i & (DIM - 1));
  int node = idx[ed];
  float coef = __expf(e2[ed] - smax[node]) / (ssum[node] + 1e-16f);
  atomicAdd(&acc[(size_t)node * DIM + d], coef * m[i]);
}

__global__ void finalize_kernel(const float* __restrict__ nf, const float* __restrict__ acc,
                                float* __restrict__ out) {
  int i = blockIdx.x * 256 + threadIdx.x;
  if (i < N_NODES * DIM) out[i] = nf[i] + acc[i] * (1.f / 3.f);
}

// ---------------------------------------------------------------------------
// Host orchestration
// ---------------------------------------------------------------------------
extern "C" void kernel_launch(void* const* d_in, const int* in_sizes, int n_in,
                              void* d_out, int out_size, void* d_ws, size_t ws_size,
                              hipStream_t stream) {
  (void)in_sizes; (void)n_in; (void)out_size; (void)ws_size;

  // --- inputs (setup_inputs dict order, params flattened in insertion order, W then b) ---
  const float* nf = (const float*)d_in[0];
  const float* ef = (const float*)d_in[1];
  const float* W_sub   = (const float*)d_in[2];   const float* b_sub   = (const float*)d_in[3];
  const float* aW_sub  = (const float*)d_in[4];   const float* ab_sub  = (const float*)d_in[5];
  const float* W_obj   = (const float*)d_in[6];   const float* b_obj   = (const float*)d_in[7];
  const float* aW_obj  = (const float*)d_in[8];   const float* ab_obj  = (const float*)d_in[9];
  const float* W_r2s   = (const float*)d_in[10];  const float* b_r2s   = (const float*)d_in[11];
  const float* aW_r2s  = (const float*)d_in[12];  const float* ab_r2s  = (const float*)d_in[13];
  const float* W_r2o   = (const float*)d_in[14];  const float* b_r2o   = (const float*)d_in[15];
  const float* aW_r2o  = (const float*)d_in[16];  const float* ab_r2o  = (const float*)d_in[17];
  const float* W_skip  = (const float*)d_in[18];  const float* b_skip  = (const float*)d_in[19];
  const float* aW_skip = (const float*)d_in[20];  const float* ab_skip = (const float*)d_in[21];
  const int* ei = (const int*)d_in[22];
  const int* es = (const int*)d_in[23];
  const int* ei0 = ei;            const int* ei1 = ei + N_EDGES;
  const int* es0 = es;            const int* es1 = es + N_EDGES;

  float* out_node = (float*)d_out;                          // [N, D]
  float* out_rel  = out_node + (size_t)N_NODES * DIM;       // [E, D]

  // --- workspace carve ---
  char* wp = (char*)d_ws;
  auto carve = [&](size_t bytes) { void* p = wp; wp += (bytes + 255) & ~(size_t)255; return p; };
  unsigned short* nf16  = (unsigned short*)carve((size_t)N_NODES * DIM * 2);
  unsigned short* w16   = (unsigned short*)carve((size_t)DIM * DIM2 * 2);     // reused per stage
  unsigned short* rel16 = (unsigned short*)carve((size_t)N_EDGES * DIM * 2);
  float* m    = (float*)carve((size_t)N_EDGES * DIM * 4);
  float* e    = (float*)carve((size_t)N_EDGES * 4);
  float* e2   = (float*)carve((size_t)N_EDGES * 4);
  float* smax = (float*)carve((size_t)N_NODES * 4);
  float* ssum = (float*)carve((size_t)N_NODES * 4);
  float* thr  = (float*)carve((size_t)N_NODES * 4);
  float* acc  = (float*)carve((size_t)N_NODES * DIM * 4);

  const dim3 gemm_grid(N_EDGES / 16, (DIM / 16) / 8);
  const int ED = N_EDGES * DIM;                  // 16.7M
  const int ed_blocks = ED / 256;
  const int attn_blocks = (N_EDGES * 32) / 256;  // one wave per edge

  // --- prep ---
  cvt_bf16_kernel<<<(N_NODES * DIM + 255) / 256, 256, 0, stream>>>(nf, nf16, N_NODES * DIM);
  zero_kernel<<<(N_NODES * DIM + 255) / 256, 256, 0, stream>>>(acc, N_NODES * DIM);

  // --- sub2rel: cat[x_i, x_j] = [nf[ei1], nf[ei0]], softmax over ei1 ---
  cvt_bf16_kernel<<<(DIM * DIM2 + 255) / 256, 256, 0, stream>>>(W_sub, w16, DIM * DIM2);
  gat_gemm_bf16_kernel<<<gemm_grid, 256, 0, stream>>>(nf16, ei1, nf16, ei0, w16, b_sub, m);
  attn_score_kernel<<<attn_blocks, 256, 0, stream>>>(m, aW_sub, ab_sub, e);
  seg_reduce_kernel<<<N_NODES, 256, 0, stream>>>(e, ei1, smax, ssum);
  rel_accum_kernel<<<ed_blocks, 256, 0, stream>>>(m, e, smax, ssum, ei1, ef, out_rel, 1);

  // --- obj2rel: cat[x_j, x_i] = [nf[ei0], nf[ei1]], softmax over ei0 ---
  cvt_bf16_kernel<<<(DIM * DIM2 + 255) / 256, 256, 0, stream>>>(W_obj, w16, DIM * DIM2);
  gat_gemm_bf16_kernel<<<gemm_grid, 256, 0, stream>>>(nf16, ei0, nf16, ei1, w16, b_obj, m);
  attn_score_kernel<<<attn_blocks, 256, 0, stream>>>(m, aW_obj, ab_obj, e);
  seg_reduce_kernel<<<N_NODES, 256, 0, stream>>>(e, ei0, smax, ssum);
  rel_accum_kernel<<<ed_blocks, 256, 0, stream>>>(m, e, smax, ssum, ei0, ef, out_rel, 0);

  // rel_emb (in d_out) -> bf16 for the next two GEMMs
  cvt_bf16_kernel<<<(ED + 255) / 256, 256, 0, stream>>>(out_rel, rel16, ED);

  // --- rel2sub: cat[x_j, rel_emb]; purifier rows grouped by ei0; softmax over ei0 ---
  cvt_bf16_kernel<<<(DIM * DIM2 + 255) / 256, 256, 0, stream>>>(W_r2s, w16, DIM * DIM2);
  gat_gemm_bf16_kernel<<<gemm_grid, 256, 0, stream>>>(nf16, ei0, rel16, nullptr, w16, b_r2s, m);
  attn_score_kernel<<<attn_blocks, 256, 0, stream>>>(m, aW_r2s, ab_r2s, e);
  topk_thresh_kernel<<<N_NODES, 256, 0, stream>>>(e, ei0, thr);
  apply_thresh_kernel<<<(N_EDGES + 255) / 256, 256, 0, stream>>>(e, ei0, thr, e2);
  seg_reduce_kernel<<<N_NODES, 256, 0, stream>>>(e2, ei0, smax, ssum);
  node_accum_kernel<<<ed_blocks, 256, 0, stream>>>(m, e2, smax, ssum, ei0, acc);

  // --- rel2obj: cat[x_i, rel_emb]; purifier cols grouped by ei1; softmax over ei1 ---
  cvt_bf16_kernel<<<(DIM * DIM2 + 255) / 256, 256, 0, stream>>>(W_r2o, w16, DIM * DIM2);
  gat_gemm_bf16_kernel<<<gemm_grid, 256, 0, stream>>>(nf16, ei1, rel16, nullptr, w16, b_r2o, m);
  attn_score_kernel<<<attn_blocks, 256, 0, stream>>>(m, aW_r2o, ab_r2o, e);
  topk_thresh_kernel<<<N_NODES, 256, 0, stream>>>(e, ei1, thr);
  apply_thresh_kernel<<<(N_EDGES + 255) / 256, 256, 0, stream>>>(e, ei1, thr, e2);
  seg_reduce_kernel<<<N_NODES, 256, 0, stream>>>(e2, ei1, smax, ssum);
  node_accum_kernel<<<ed_blocks, 256, 0, stream>>>(m, e2, smax, ssum, ei1, acc);

  // --- skip: cat[nf[es1], nf[es0]]; purifier cols grouped by es1; softmax over es1 ---
  cvt_bf16_kernel<<<(DIM * DIM2 + 255) / 256, 256, 0, stream>>>(W_skip, w16, DIM * DIM2);
  gat_gemm_bf16_kernel<<<gemm_grid, 256, 0, stream>>>(nf16, es1, nf16, es0, w16, b_skip, m);
  attn_score_kernel<<<attn_blocks, 256, 0, stream>>>(m, aW_skip, ab_skip, e);
  topk_thresh_kernel<<<N_NODES, 256, 0, stream>>>(e, es1, thr);
  apply_thresh_kernel<<<(N_EDGES + 255) / 256, 256, 0, stream>>>(e, es1, thr, e2);
  seg_reduce_kernel<<<N_NODES, 256, 0, stream>>>(e2, es1, smax, ssum);
  node_accum_kernel<<<ed_blocks, 256, 0, stream>>>(m, e2, smax, ssum, es1, acc);

  // --- node_emb = nf + (sum of three aggregations)/3 ---
  finalize_kernel<<<(N_NODES * DIM + 255) / 256, 256, 0, stream>>>(nf, acc, out_node);
}
